// Attention_72456098284196
// MI455X (gfx1250) — matrix operand order repaired and checked by reading, not compiled, not run on hardware
//
#include <hip/hip_runtime.h>

typedef __attribute__((ext_vector_type(16))) _Float16 v16h;
typedef __attribute__((ext_vector_type(8)))  _Float16 v8h;
typedef __attribute__((ext_vector_type(8)))  float    v8f;

#define BB 2
#define TT 2048
#define CC 1024
#define HH 16
#define DD 64

static __device__ __forceinline__ v16h cat16(v8h lo, v8h hi) {
    return __builtin_shufflevector(lo, hi, 0,1,2,3,4,5,6,7,8,9,10,11,12,13,14,15);
}

static __device__ __forceinline__ v8f wmma_f16(v16h a, v16h b, v8f c) {
    // v_wmma_f32_16x16x32_f16: D = A(16x32 f16) * B(32x16 f16) + C(16x16 f32)
    return __builtin_amdgcn_wmma_f32_16x16x32_f16(false, a, false, b, (short)0, c, false, false);
}

// CDNA5 async copy: global -> LDS, 16B per lane, tracked by ASYNCcnt.
// LDS offset = low 32 bits of the generic pointer to __shared__ memory.
static __device__ __forceinline__ void async_ld_b128(void* lds, const void* g) {
    unsigned l = (unsigned)(unsigned long long)lds;
    unsigned long long ga = (unsigned long long)g;
    asm volatile("global_load_async_to_lds_b128 %0, %1, off"
                 :: "v"(l), "v"(ga) : "memory");
}
static __device__ __forceinline__ void wait_async0() {
    asm volatile("s_wait_asynccnt 0" ::: "memory");
}

// Reduce across the 16-lane half-groups of a wave32 (N-dimension of C/D frags)
static __device__ __forceinline__ float half_group_max(float x) {
    x = fmaxf(x, __shfl_xor(x, 1, 32));
    x = fmaxf(x, __shfl_xor(x, 2, 32));
    x = fmaxf(x, __shfl_xor(x, 4, 32));
    x = fmaxf(x, __shfl_xor(x, 8, 32));
    return x;
}
static __device__ __forceinline__ float half_group_sum(float x) {
    x += __shfl_xor(x, 1, 32);
    x += __shfl_xor(x, 2, 32);
    x += __shfl_xor(x, 4, 32);
    x += __shfl_xor(x, 8, 32);
    return x;
}

// ---------------------------------------------------------------------------
// f32 -> f16 conversion (bandwidth trivial; enables f16 WMMA everywhere)
// ---------------------------------------------------------------------------
__global__ __launch_bounds__(256) void cvt_f32_f16(const float* __restrict__ src,
                                                   _Float16* __restrict__ dst, int n8) {
    int i = blockIdx.x * 256 + threadIdx.x;
    if (i < n8) {
        float4 f0 = *(const float4*)(src + (size_t)i * 8);
        float4 f1 = *(const float4*)(src + (size_t)i * 8 + 4);
        v8h o;
        o[0] = (_Float16)f0.x; o[1] = (_Float16)f0.y;
        o[2] = (_Float16)f0.z; o[3] = (_Float16)f0.w;
        o[4] = (_Float16)f1.x; o[5] = (_Float16)f1.y;
        o[6] = (_Float16)f1.z; o[7] = (_Float16)f1.w;
        *(v8h*)(dst + (size_t)i * 8) = o;
    }
}

// ---------------------------------------------------------------------------
// Fused projection:  dst[b][h][t][d] = f16( x @ W^T + bias, (rope?) )
// out[m][n] = sum_k X[m][k] * W[n][k]  -> B-matrix column n == W row n.
// Grid (32, 16): 128x64 output tile per 256-thread block (8 waves, wave w
// owns rows w*16..w*16+15, all 64 cols => 4 WMMA n-tiles). A/B tiles are
// double-buffered in LDS, filled with async global->LDS loads issued one
// k-step ahead (s_wait_asynccnt 0 + barrier at top of each step).
// N-block == 64 == head size, so blockIdx.y is the head index.
// ---------------------------------------------------------------------------
__global__ __launch_bounds__(256) void qkv_proj_rope(
        const _Float16* __restrict__ X, const _Float16* __restrict__ W,
        const float* __restrict__ bias, const int* __restrict__ pos,
        _Float16* __restrict__ dst, int do_rope)
{
    __shared__ __align__(16) _Float16 As[2][128 * 32];
    __shared__ __align__(16) _Float16 Bs[2][64 * 32];

    const int tid   = threadIdx.x;
    const int lane  = tid & 31;
    const int wave  = tid >> 5;
    const int lhalf = lane & 15;   // N index inside a 16-wide tile
    const int lhi   = lane >> 4;   // which half-group
    const int mb = blockIdx.x;     // row block (128 rows of B*T)
    const int nb = blockIdx.y;     // col block (64 cols) == head index

    // ---- prologue: async-stage k-tile 0 into buffer 0 ----
    {
        #pragma unroll
        for (int r = 0; r < 2; ++r) {                // A: 512 chunks / 256 thr
            int i = tid + r * 256;
            int row = i >> 2, col = (i & 3) * 8;
            async_ld_b128(&As[0][row * 32 + col],
                          &X[(size_t)(mb * 128 + row) * CC + col]);
        }
        int row = tid >> 2, col = (tid & 3) * 8;     // B: 256 chunks / 256 thr
        async_ld_b128(&Bs[0][row * 32 + col],
                      &W[(size_t)(nb * 64 + row) * CC + col]);
    }

    v8f acc[4] = {};

    for (int kb = 0; kb < CC; kb += 32) {
        const int cur = (kb >> 5) & 1;
        wait_async0();
        __syncthreads();                             // tile kb resident
        if (kb + 32 < CC) {                          // issue tile kb+32
            const int nxt = cur ^ 1;
            #pragma unroll
            for (int r = 0; r < 2; ++r) {
                int i = tid + r * 256;
                int row = i >> 2, col = (i & 3) * 8;
                async_ld_b128(&As[nxt][row * 32 + col],
                              &X[(size_t)(mb * 128 + row) * CC + kb + 32 + col]);
            }
            int row = tid >> 2, col = (tid & 3) * 8;
            async_ld_b128(&Bs[nxt][row * 32 + col],
                          &W[(size_t)(nb * 64 + row) * CC + kb + 32 + col]);
        }

        // Batch all fragment loads first, then run the WMMA chain, so LDS
        // returns overlap the matrix pipe instead of one wait per WMMA.
        const int am = wave * 16 + lhalf;
        v16h a = cat16(*(v8h*)&As[cur][am * 32 + lhi * 8],
                       *(v8h*)&As[cur][am * 32 + 16 + lhi * 8]);
        v16h bfr[4];
        #pragma unroll
        for (int nt = 0; nt < 4; ++nt) {
            const int bn = nt * 16 + lhalf;
            bfr[nt] = cat16(*(v8h*)&Bs[cur][bn * 32 + lhi * 8],
                            *(v8h*)&Bs[cur][bn * 32 + 16 + lhi * 8]);
        }
        #pragma unroll
        for (int nt = 0; nt < 4; ++nt) acc[nt] = wmma_f16(a, bfr[nt], acc[nt]);
    }

    // Epilogue: bias + RoPE (first 32 channels of the head) + f16 store to
    // dst[((b*H + nb)*T + t)*64 + d].
    #pragma unroll
    for (int nt = 0; nt < 4; ++nt) {
        const int d    = nt * 16 + lhalf;       // channel within head
        const float bval = bias[nb * 64 + d];
        // inv_freq = 8192^(-band/15) = 2^(-13*band/15), band = d>>1 (d < 32)
        const float invf = exp2f(-13.0f * (float)(d >> 1) / 15.0f);
        #pragma unroll
        for (int v = 0; v < 8; ++v) {
            const int row = mb * 128 + wave * 16 + v + 8 * lhi;  // 0..4095
            const int b   = row >> 11;          // row / T
            const int t   = row & (TT - 1);
            float val = acc[nt][v] + bval;
            float outv = val;
            if (do_rope && nt < 2) {            // uniform branch: d < 32
                float p  = (float)pos[b * TT + t];
                float r  = p * invf;
                float cs = __cosf(r), sn = __sinf(r);
                float partner = __shfl_xor(val, 1, 32);   // channel d^1, same row
                outv = (d & 1) ? (val * cs + partner * sn)
                               : (val * cs - partner * sn);
            }
            dst[((size_t)(b * HH + nb) * TT + t) * DD + d] = (_Float16)outv;
        }
    }
}

// ---------------------------------------------------------------------------
// Flash attention, one (b,h) head + 64 query rows per 128-thread workgroup.
// Wave w owns query rows w*16..w*16+15. KV processed in 64-key tiles:
// K tiles double-buffered via async global->LDS; V tiles prefetched into
// registers one tile ahead and stored transposed into a double-buffered
// [d][key] LDS image (B-fragments become contiguous ds_load_b128).
// S = Q K^T (8 WMMA/wave), online softmax in C/D fragment layout, P restaged
// through per-wave LDS into A-fragment layout, O += P V (8 WMMA/wave).
// ---------------------------------------------------------------------------
__global__ __launch_bounds__(128) void flash_attn(
        const _Float16* __restrict__ Q, const _Float16* __restrict__ K,
        const _Float16* __restrict__ V, float* __restrict__ out)
{
    __shared__ __align__(16) _Float16 Ks[2][64 * 64];    // [key][d]
    __shared__ __align__(16) _Float16 Vt[2][64 * 64];    // [d][key] (transposed)
    __shared__ __align__(16) _Float16 Ps[4][16 * 64];    // per-wave P staging

    const int tid   = threadIdx.x;
    const int lane  = tid & 31;
    const int wave  = tid >> 5;
    const int lhalf = lane & 15;
    const int lhi   = lane >> 4;
    const int qb = blockIdx.x;                 // query block: 0..31
    const int bh = blockIdx.y;                 // b*H + h:     0..31
    const size_t headbase = (size_t)bh * TT * DD;

    // Preload this wave's Q fragments (16 rows x 64 dims, two K=32 chunks).
    const int qrow = qb * 64 + wave * 16 + lhalf;
    const _Float16* qp = Q + headbase + (size_t)qrow * DD;
    v16h qf[2];
    #pragma unroll
    for (int c = 0; c < 2; ++c)
        qf[c] = cat16(*(const v8h*)&qp[c * 32 + lhi * 8],
                      *(const v8h*)&qp[c * 32 + 16 + lhi * 8]);

    // ---- prologue: stage tile 0 (K async, V via registers, transposed) ----
    #pragma unroll
    for (int r = 0; r < 4; ++r) {              // K: 512 chunks / 128 threads
        int i = tid + r * 128;
        async_ld_b128(&Ks[0][i * 8], &K[headbase + (size_t)i * 8]);
    }
    {
        v8h vreg[4];
        #pragma unroll
        for (int r = 0; r < 4; ++r) {
            int i = tid + r * 128;
            int t = i >> 3, d0 = (i & 7) * 8;
            vreg[r] = *(const v8h*)&V[headbase + (size_t)t * DD + d0];
        }
        #pragma unroll
        for (int r = 0; r < 4; ++r) {
            int i = tid + r * 128;
            int t = i >> 3, d0 = (i & 7) * 8;
            #pragma unroll
            for (int j = 0; j < 8; ++j) Vt[0][(d0 + j) * 64 + t] = vreg[r][j];
        }
    }

    v8f oacc[4] = {};
    float m_i[8], l_i[8];
    #pragma unroll
    for (int v = 0; v < 8; ++v) { m_i[v] = -__builtin_inff(); l_i[v] = 0.0f; }

    const float scale = 0.125f;                // 1/sqrt(64)

    for (int kb = 0; kb < TT; kb += 64) {
        const int cur = (kb >> 6) & 1;
        const int nxt = cur ^ 1;
        const bool have_next = (kb + 64) < TT; // uniform

        wait_async0();
        __syncthreads();                       // K[cur]/Vt[cur] resident

        v8h vreg[4];
        if (have_next) {
            // issue next K tile (async) and start next V tile global loads
            #pragma unroll
            for (int r = 0; r < 4; ++r) {
                int i = tid + r * 128;
                async_ld_b128(&Ks[nxt][i * 8],
                              &K[headbase + (size_t)(kb + 64) * DD + i * 8]);
            }
            #pragma unroll
            for (int r = 0; r < 4; ++r) {
                int i = tid + r * 128;
                int t = i >> 3, d0 = (i & 7) * 8;
                vreg[r] = *(const v8h*)&V[headbase + (size_t)(kb + 64 + t) * DD + d0];
            }
        }

        // ---- S = Q K^T : 4 n-tiles of 16 keys, K-dim = 64 in 2 chunks ----
        v8f sacc[4] = {};
        #pragma unroll
        for (int c = 0; c < 2; ++c) {
            v16h bfr[4];
            #pragma unroll
            for (int nt = 0; nt < 4; ++nt) {
                const int kt = nt * 16 + lhalf;   // B column == key row of Ks
                bfr[nt] = cat16(*(v8h*)&Ks[cur][kt * 64 + c * 32 + lhi * 8],
                                *(v8h*)&Ks[cur][kt * 64 + c * 32 + 16 + lhi * 8]);
            }
            #pragma unroll
            for (int nt = 0; nt < 4; ++nt)
                sacc[nt] = wmma_f16(qf[c], bfr[nt], sacc[nt]);
        }

        // ---- online softmax over the 64 keys of this tile ----
        float alpha[8];
        #pragma unroll
        for (int v = 0; v < 8; ++v) {
            float mx = fmaxf(fmaxf(sacc[0][v], sacc[1][v]),
                             fmaxf(sacc[2][v], sacc[3][v])) * scale;
            mx = half_group_max(mx);
            const float mnew = fmaxf(m_i[v], mx);
            alpha[v] = __expf(m_i[v] - mnew);
            float rs = 0.0f;
            #pragma unroll
            for (int nt = 0; nt < 4; ++nt) {
                float p = __expf(sacc[nt][v] * scale - mnew);
                rs += p;
                // C/D layout -> row-major P in this wave's staging buffer
                Ps[wave][(v + 8 * lhi) * 64 + nt * 16 + lhalf] = (_Float16)p;
            }
            rs = half_group_sum(rs);
            l_i[v] = l_i[v] * alpha[v] + rs;
            m_i[v] = mnew;
        }
        #pragma unroll
        for (int nt = 0; nt < 4; ++nt)
            #pragma unroll
            for (int v = 0; v < 8; ++v) oacc[nt][v] *= alpha[v];

        if (have_next) {                       // store next V tile transposed
            #pragma unroll
            for (int r = 0; r < 4; ++r) {
                int i = tid + r * 128;
                int t = i >> 3, d0 = (i & 7) * 8;
                #pragma unroll
                for (int j = 0; j < 8; ++j) Vt[nxt][(d0 + j) * 64 + t] = vreg[r][j];
            }
        }

        __syncthreads();                       // P visible (Vt[nxt] covered by
                                               // next iteration's top barrier)

        // ---- O += P * V : A = P (16x64), B = Vt rows are B columns ----
        #pragma unroll
        for (int c = 0; c < 2; ++c) {
            v16h pa = cat16(*(v8h*)&Ps[wave][lhalf * 64 + c * 32 + lhi * 8],
                            *(v8h*)&Ps[wave][lhalf * 64 + c * 32 + 16 + lhi * 8]);
            v16h bfr[4];
            #pragma unroll
            for (int nt = 0; nt < 4; ++nt) {
                const int dn = nt * 16 + lhalf;   // output channel d
                bfr[nt] = cat16(*(v8h*)&Vt[cur][dn * 64 + c * 32 + lhi * 8],
                                *(v8h*)&Vt[cur][dn * 64 + c * 32 + 16 + lhi * 8]);
            }
            #pragma unroll
            for (int nt = 0; nt < 4; ++nt)
                oacc[nt] = wmma_f16(pa, bfr[nt], oacc[nt]);
        }
    }

    // Epilogue: normalize and store fp32 out[b][t][h*64+d].
    const int b = bh >> 4, h = bh & 15;
    #pragma unroll
    for (int nt = 0; nt < 4; ++nt) {
        const int d = nt * 16 + lhalf;
        #pragma unroll
        for (int v = 0; v < 8; ++v) {
            const int t = qb * 64 + wave * 16 + v + 8 * lhi;
            out[((size_t)(b * TT + t) * HH + h) * DD + d] = oacc[nt][v] / l_i[v];
        }
    }
}

// ---------------------------------------------------------------------------
extern "C" void kernel_launch(void* const* d_in, const int* in_sizes, int n_in,
                              void* d_out, int out_size, void* d_ws, size_t ws_size,
                              hipStream_t stream) {
    (void)in_sizes; (void)n_in; (void)out_size; (void)ws_size;
    const float* x_q   = (const float*)d_in[0];
    const float* x_kv  = (const float*)d_in[1];
    const int*   q_pos = (const int*)  d_in[2];
    const int*   kv_pos= (const int*)  d_in[3];
    const float* Wq    = (const float*)d_in[4];
    const float* bq    = (const float*)d_in[5];
    const float* Wk    = (const float*)d_in[6];
    const float* bk    = (const float*)d_in[7];
    const float* Wv    = (const float*)d_in[8];
    const float* bv    = (const float*)d_in[9];
    float* out = (float*)d_out;

    const size_t XN = (size_t)BB * TT * CC;   // 4 Mi elements
    const size_t WN = (size_t)CC * CC;        // 1 Mi elements
    const size_t QN = (size_t)BB * HH * TT * DD;

    _Float16* ws    = (_Float16*)d_ws;
    _Float16* xq16  = ws;
    _Float16* xkv16 = xq16  + XN;
    _Float16* wq16  = xkv16 + XN;
    _Float16* wk16  = wq16  + WN;
    _Float16* wv16  = wk16  + WN;
    _Float16* q16   = wv16  + WN;
    _Float16* k16   = q16   + QN;
    _Float16* v16   = k16   + QN;

    cvt_f32_f16<<<dim3((unsigned)(XN / 8 / 256)), dim3(256), 0, stream>>>(x_q,  xq16,  (int)(XN / 8));
    cvt_f32_f16<<<dim3((unsigned)(XN / 8 / 256)), dim3(256), 0, stream>>>(x_kv, xkv16, (int)(XN / 8));
    cvt_f32_f16<<<dim3((unsigned)(WN / 8 / 256)), dim3(256), 0, stream>>>(Wq,   wq16,  (int)(WN / 8));
    cvt_f32_f16<<<dim3((unsigned)(WN / 8 / 256)), dim3(256), 0, stream>>>(Wk,   wk16,  (int)(WN / 8));
    cvt_f32_f16<<<dim3((unsigned)(WN / 8 / 256)), dim3(256), 0, stream>>>(Wv,   wv16,  (int)(WN / 8));

    dim3 gg(32, 16), gb(256);
    qkv_proj_rope<<<gg, gb, 0, stream>>>(xq16,  wq16, bq, q_pos,  q16, 1);
    qkv_proj_rope<<<gg, gb, 0, stream>>>(xkv16, wk16, bk, kv_pos, k16, 1);
    qkv_proj_rope<<<gg, gb, 0, stream>>>(xkv16, wv16, bv, nullptr, v16, 0);

    flash_attn<<<dim3(32, 32), dim3(128), 0, stream>>>(q16, k16, v16, out);
}
